// Mamba1_25503515803772
// MI455X (gfx1250) — matrix operand order repaired and checked by reading, not compiled, not run on hardware
//
#include <hip/hip_runtime.h>
#include <hip/hip_bf16.h>
#include <math.h>

#define BATCH   2
#define SEQ     1024
#define MROWS   (BATCH * SEQ)      // 2048
#define D_MODEL 768
#define D_INNER 1536
#define DT_RANK 48
#define DT_PAD  64
#define N_STATE 16
#define D_CONV  4
#define N_LAYER 2
#define VOCAB   50304
#define XP_OUT  (DT_RANK + 2 * N_STATE)   // 80
#define XP_PAD  128
#define M_CHUNKS 8
#define M_CHUNK (MROWS / M_CHUNKS)        // 256

typedef __attribute__((ext_vector_type(16))) __bf16        v16bf;
typedef __attribute__((ext_vector_type(8)))  float         v8f;
typedef __attribute__((ext_vector_type(4)))  unsigned int  u32x4;
typedef __attribute__((ext_vector_type(4)))  int           v4i;

union FragU { u32x4 u[2]; v16bf v; };

#if defined(__has_builtin)
#  if __has_builtin(__builtin_amdgcn_global_load_async_to_lds_b128)
#    define HAVE_ASYNC_LDS 1
#  endif
#endif
#ifndef HAVE_ASYNC_LDS
#  define HAVE_ASYNC_LDS 0
#endif

#if HAVE_ASYNC_LDS
typedef __attribute__((address_space(1))) v4i glb_v4i;
typedef __attribute__((address_space(3))) v4i lds_v4i;
#endif

__device__ __forceinline__ void async_wait0() {
#if defined(__has_builtin) && __has_builtin(__builtin_amdgcn_s_wait_asynccnt)
    __builtin_amdgcn_s_wait_asynccnt(0);
#else
    asm volatile("s_wait_asynccnt 0" ::: "memory");
#endif
}

__device__ __forceinline__ unsigned short f2bf(float f) {
    union { float f; unsigned u; } c; c.f = f;
    unsigned r = c.u + 0x7FFFu + ((c.u >> 16) & 1u);   // round-to-nearest-even
    return (unsigned short)(r >> 16);
}
__device__ __forceinline__ float silu_f(float x)     { return x / (1.f + __expf(-x)); }
__device__ __forceinline__ float softplus_f(float x) { return (x > 20.f) ? x : log1pf(__expf(x)); }

// ---------------------------------------------------------------- embedding
__global__ void mamba_embed(const int* __restrict__ tokens,
                            const float* __restrict__ wte,
                            float* __restrict__ x) {
    int row = blockIdx.x;                     // b*SEQ + l
    int tok = tokens[row];
    const float* src = wte + (size_t)tok * D_MODEL;
    float* dst = x + (size_t)row * D_MODEL;
    for (int j = threadIdx.x; j < D_MODEL; j += blockDim.x) dst[j] = src[j];
}

// ------------------------------------------------- fp32 -> bf16 convert+pad
__global__ void mamba_cvt_pad(const float* __restrict__ src,
                              unsigned short* __restrict__ dst,
                              int Rd, int Kd, int Rs, int Ks, int srcStride) {
    int idx = blockIdx.x * blockDim.x + threadIdx.x;
    int total = Rd * Kd;
    if (idx >= total) return;
    int r = idx / Kd, k = idx - r * Kd;
    float v = (r < Rs && k < Ks) ? src[(size_t)r * srcStride + k] : 0.f;
    dst[idx] = f2bf(v);
}

// ------------------------------------------------------------------ rmsnorm
__global__ void mamba_rmsnorm(const float* __restrict__ x,
                              unsigned short* __restrict__ out, int D) {
    __shared__ float red[256];
    int row = blockIdx.x, tid = threadIdx.x;
    const float* px = x + (size_t)row * D;
    float s = 0.f;
    for (int j = tid; j < D; j += 256) { float v = px[j]; s += v * v; }
    red[tid] = s; __syncthreads();
    for (int off = 128; off > 0; off >>= 1) {
        if (tid < off) red[tid] += red[tid + off];
        __syncthreads();
    }
    float rs = rsqrtf(red[0] / (float)D + 1e-6f);
    unsigned short* po = out + (size_t)row * D;
    for (int j = tid; j < D; j += 256) po[j] = f2bf(px[j] * rs);
}

// --------------------------------------------- WMMA GEMM, LDS-resident weights
// C[M,N] (f32) = A_bf16[M,K] @ W_bf16[N,K]^T, both K-contiguous row-major.
// Each block pins an NW-column weight strip (NW*K bf16) in LDS (async global->LDS
// when available), then loops over M-tiles of its M-chunk with weights resident.
// MODE 0: plain   MODE 1: C += residual aux[idx]   MODE 2: softplus(+aux[col])
template <int MODE, int NW>
__global__ __launch_bounds__(256) void mamba_gemm_wmma(
        const unsigned short* __restrict__ A,
        const unsigned short* __restrict__ W,
        float* __restrict__ C, int M, int N, int K,
        const float* __restrict__ aux, int mChunk) {
    extern __shared__ unsigned short sW[];            // NW * K bf16
    constexpr int WC = NW / 32;                       // wave columns (2 or 4)
    constexpr int WR = 8 / WC;                        // wave rows    (4 or 2)
    constexpr int BM = WR * 32;                       // M-tile per iteration
    int tid  = threadIdx.x;
    int lane = tid & 31, wv = tid >> 5;               // 8 waves (wave32)
    int wn = wv % WC, wm = wv / WC;
    int nStrip = blockIdx.x * NW;
    int l15 = lane & 15, lhi = lane >> 4;

    // ---- stage the weight strip into LDS (row-major NW x K, same as global)
    {
        const unsigned short* Wg = W + (size_t)nStrip * K;
        int total = NW * K;
        for (int e = tid * 8; e < total; e += 256 * 8) {
#if HAVE_ASYNC_LDS
            __builtin_amdgcn_global_load_async_to_lds_b128(
                (glb_v4i*)(Wg + e), (lds_v4i*)(sW + e), 0, 0);
#else
            *(u32x4*)(sW + e) = *(const u32x4*)(Wg + e);
#endif
        }
#if HAVE_ASYNC_LDS
        async_wait0();
#endif
    }
    __syncthreads();

    // B fragment LDS bases: lane holds col n = base + l15; lanes 16-31 take K+8 half
    const unsigned short* sB0 = sW + (size_t)(wn * 32 + l15) * K + lhi * 8;
    const unsigned short* sB1 = sB0 + (size_t)16 * K;
    int akb = lhi * 8;

    int mBeg = blockIdx.y * mChunk;
    for (int m0 = mBeg; m0 < mBeg + mChunk; m0 += BM) {
        // A fragment bases: lane holds row m = base + l15; lanes 16-31 take K+8 half
        const unsigned short* aRow0 = A + (size_t)(m0 + wm * 32 + l15) * K + akb;
        const unsigned short* aRow1 = aRow0 + (size_t)16 * K;

        v8f acc00 = {}, acc01 = {}, acc10 = {}, acc11 = {};
        for (int k0 = 0; k0 < K; k0 += 32) {
            FragU a0, a1, b0, b1;
            const unsigned short* p;
            p = aRow0 + k0; a0.u[0] = *(const u32x4*)p; a0.u[1] = *(const u32x4*)(p + 16);
            p = aRow1 + k0; a1.u[0] = *(const u32x4*)p; a1.u[1] = *(const u32x4*)(p + 16);
            p = sB0   + k0; b0.u[0] = *(const u32x4*)p; b0.u[1] = *(const u32x4*)(p + 16);
            p = sB1   + k0; b1.u[0] = *(const u32x4*)p; b1.u[1] = *(const u32x4*)(p + 16);
            acc00 = __builtin_amdgcn_wmma_f32_16x16x32_bf16(false, a0.v, false, b0.v, (short)0, acc00, false, false);
            acc01 = __builtin_amdgcn_wmma_f32_16x16x32_bf16(false, a0.v, false, b1.v, (short)0, acc01, false, false);
            acc10 = __builtin_amdgcn_wmma_f32_16x16x32_bf16(false, a1.v, false, b0.v, (short)0, acc10, false, false);
            acc11 = __builtin_amdgcn_wmma_f32_16x16x32_bf16(false, a1.v, false, b1.v, (short)0, acc11, false, false);
        }

        // C/D layout: lanes 0-15 rows v+0..7, lanes 16-31 rows v+8..15; col = l15
        v8f* accs[2][2] = { { &acc00, &acc01 }, { &acc10, &acc11 } };
        #pragma unroll
        for (int mi = 0; mi < 2; ++mi) {
            #pragma unroll
            for (int ni = 0; ni < 2; ++ni) {
                int col  = nStrip + wn * 32 + ni * 16 + l15;
                int row0 = m0 + wm * 32 + mi * 16 + lhi * 8;
                if (col < N) {
                    v8f a = *accs[mi][ni];
                    #pragma unroll
                    for (int v = 0; v < 8; ++v) {
                        size_t idx = (size_t)(row0 + v) * N + col;
                        float val = a[v];
                        if (MODE == 1) val += aux[idx];
                        if (MODE == 2) val = softplus_f(val + aux[col]);
                        C[idx] = val;
                    }
                }
            }
        }
    }
}

// ------------------------------------------- depthwise causal conv4 + SiLU
__global__ void mamba_conv_silu(const float* __restrict__ xz,
                                const float* __restrict__ cw,
                                const float* __restrict__ cb,
                                unsigned short* __restrict__ xc_bf,
                                float* __restrict__ xc_f) {
    int d = blockIdx.x * 256 + threadIdx.x;   // channel
    int l = blockIdx.y, b = blockIdx.z;
    float acc = cb[d];
    #pragma unroll
    for (int j = 0; j < D_CONV; ++j) {
        int lp = l - (D_CONV - 1) + j;
        if (lp >= 0)
            acc += cw[d * D_CONV + j] * xz[((size_t)(b * SEQ + lp)) * (2 * D_INNER) + d];
    }
    float s = silu_f(acc);
    size_t idx = (size_t)(b * SEQ + l) * D_INNER + d;
    xc_f[idx]  = s;
    xc_bf[idx] = f2bf(s);
}

// -------------------------------------------------------- selective scan
__global__ void mamba_scan(const float* __restrict__ dt,
                           const float* __restrict__ xcf,
                           const float* __restrict__ xdbl,   // (MROWS, 80)
                           const float* __restrict__ A_log,
                           const float* __restrict__ Dvec,
                           float* __restrict__ ys) {
    int b = blockIdx.y;
    int d = blockIdx.x * 256 + threadIdx.x;
    float a[N_STATE], h[N_STATE];
    #pragma unroll
    for (int n = 0; n < N_STATE; ++n) {
        a[n] = -__expf(A_log[(size_t)d * N_STATE + n]);
        h[n] = 0.f;
    }
    float Dp = Dvec[d];
    __shared__ float sB[N_STATE], sC[N_STATE];
    for (int t = 0; t < SEQ; ++t) {
        int row = b * SEQ + t;
        __syncthreads();
        if (threadIdx.x < 32) {
            float v = xdbl[(size_t)row * XP_OUT + DT_RANK + threadIdx.x];
            if (threadIdx.x < 16) sB[threadIdx.x] = v;
            else                  sC[threadIdx.x - 16] = v;
        }
        __syncthreads();
        float dtv = dt[(size_t)row * D_INNER + d];
        float xv  = xcf[(size_t)row * D_INNER + d];
        float y = 0.f;
        #pragma unroll
        for (int n = 0; n < N_STATE; ++n) {
            h[n] = __expf(dtv * a[n]) * h[n] + dtv * sB[n] * xv;
            y += h[n] * sC[n];
        }
        ys[(size_t)row * D_INNER + d] = y + Dp * xv;
    }
}

// --------------------------------------------------------- y * silu(z)
__global__ void mamba_gate(const float* __restrict__ ys,
                           const float* __restrict__ xz,
                           unsigned short* __restrict__ yz_bf) {
    int idx = blockIdx.x * 256 + threadIdx.x;
    if (idx >= MROWS * D_INNER) return;
    int row = idx / D_INNER, d = idx - row * D_INNER;
    float z = xz[(size_t)row * (2 * D_INNER) + D_INNER + d];
    yz_bf[idx] = f2bf(ys[idx] * silu_f(z));
}

// ===========================================================================
extern "C" void kernel_launch(void* const* d_in, const int* in_sizes, int n_in,
                              void* d_out, int out_size, void* d_ws, size_t ws_size,
                              hipStream_t stream) {
    const int*   tokens    = (const int*)  d_in[0];
    const float* wte       = (const float*)d_in[1];
    const float* in_proj_w = (const float*)d_in[2];   // (2, 3072, 768)
    const float* conv_w    = (const float*)d_in[3];   // (2, 1536, 1, 4)
    const float* conv_b    = (const float*)d_in[4];   // (2, 1536)
    const float* x_proj_w  = (const float*)d_in[5];   // (2, 80, 1536)
    const float* dt_proj_w = (const float*)d_in[6];   // (2, 1536, 48)
    const float* dt_proj_b = (const float*)d_in[7];   // (2, 1536)
    const float* out_proj_w= (const float*)d_in[8];   // (2, 768, 1536)
    const float* A_log     = (const float*)d_in[9];   // (2, 1536, 16)
    const float* Dvec      = (const float*)d_in[10];  // (2, 1536)
    const float* lm_head_w = (const float*)d_in[11];  // (50304, 768)
    float* out = (float*)d_out;

    // ---- workspace bump allocator (256B aligned)
    char* cur = (char*)d_ws;
    auto alloc = [&](size_t bytes) -> void* {
        void* p = cur; cur += (bytes + 255) & ~(size_t)255; return p;
    };
    float*          x       = (float*)         alloc((size_t)MROWS * D_MODEL * 4);
    unsigned short* xn      = (unsigned short*)alloc((size_t)MROWS * D_MODEL * 2);
    float*          xz      = (float*)         alloc((size_t)MROWS * 2 * D_INNER * 4);
    unsigned short* xc_bf   = (unsigned short*)alloc((size_t)MROWS * D_INNER * 2);
    float*          xc_f    = (float*)         alloc((size_t)MROWS * D_INNER * 4);
    float*          xdbl    = (float*)         alloc((size_t)MROWS * XP_OUT * 4);
    unsigned short* dtpad   = (unsigned short*)alloc((size_t)MROWS * DT_PAD * 2);
    float*          dtbuf   = (float*)         alloc((size_t)MROWS * D_INNER * 4);
    float*          ys      = (float*)         alloc((size_t)MROWS * D_INNER * 4);
    unsigned short* yz_bf   = (unsigned short*)alloc((size_t)MROWS * D_INNER * 2);
    unsigned short* w_in    = (unsigned short*)alloc((size_t)2 * D_INNER * D_MODEL * 2);
    unsigned short* w_xp    = (unsigned short*)alloc((size_t)XP_PAD * D_INNER * 2);
    unsigned short* w_dt    = (unsigned short*)alloc((size_t)D_INNER * DT_PAD * 2);
    unsigned short* w_out   = (unsigned short*)alloc((size_t)D_MODEL * D_INNER * 2);
    unsigned short* w_lm    = (unsigned short*)alloc((size_t)VOCAB * D_MODEL * 2);

    auto cvt = [&](const float* src, unsigned short* dst, int Rd, int Kd,
                   int Rs, int Ks, int str) {
        int total = Rd * Kd;
        mamba_cvt_pad<<<(total + 255) / 256, 256, 0, stream>>>(src, dst, Rd, Kd, Rs, Ks, str);
    };

    // ---- embedding
    mamba_embed<<<MROWS, 256, 0, stream>>>(tokens, wte, x);

    for (int i = 0; i < N_LAYER; ++i) {
        // rmsnorm -> bf16
        mamba_rmsnorm<<<MROWS, 256, 0, stream>>>(x, xn, D_MODEL);

        // in_proj: xz = xn @ in_w^T   (2048 x 3072, K=768) — NW=128, LDS 192KB
        cvt(in_proj_w + (size_t)i * 2 * D_INNER * D_MODEL, w_in,
            2 * D_INNER, D_MODEL, 2 * D_INNER, D_MODEL, D_MODEL);
        mamba_gemm_wmma<0, 128><<<dim3(2 * D_INNER / 128, M_CHUNKS), 256,
                                  128 * D_MODEL * 2, stream>>>(
            xn, w_in, xz, MROWS, 2 * D_INNER, D_MODEL, nullptr, M_CHUNK);

        // depthwise conv + silu
        mamba_conv_silu<<<dim3(D_INNER / 256, SEQ, BATCH), 256, 0, stream>>>(
            xz, conv_w + (size_t)i * D_INNER * D_CONV, conv_b + (size_t)i * D_INNER,
            xc_bf, xc_f);

        // x_proj: xdbl = xc @ xp^T (2048 x 80, K=1536) — NW=64, W padded to 128 rows
        cvt(x_proj_w + (size_t)i * XP_OUT * D_INNER, w_xp,
            XP_PAD, D_INNER, XP_OUT, D_INNER, D_INNER);
        mamba_gemm_wmma<0, 64><<<dim3(XP_PAD / 64, M_CHUNKS), 256,
                                 64 * D_INNER * 2, stream>>>(
            xc_bf, w_xp, xdbl, MROWS, XP_OUT, D_INNER, nullptr, M_CHUNK);

        // dt slice (cols 0..47 of xdbl) -> bf16, K padded 48->64
        cvt(xdbl, dtpad, MROWS, DT_PAD, MROWS, DT_RANK, XP_OUT);

        // dt_proj + bias + softplus   (2048 x 1536, K=64) — NW=128, LDS 16KB
        cvt(dt_proj_w + (size_t)i * D_INNER * DT_RANK, w_dt,
            D_INNER, DT_PAD, D_INNER, DT_RANK, DT_RANK);
        mamba_gemm_wmma<2, 128><<<dim3(D_INNER / 128, M_CHUNKS), 256,
                                  128 * DT_PAD * 2, stream>>>(
            dtpad, w_dt, dtbuf, MROWS, D_INNER, DT_PAD,
            dt_proj_b + (size_t)i * D_INNER, M_CHUNK);

        // selective scan
        mamba_scan<<<dim3(D_INNER / 256, BATCH), 256, 0, stream>>>(
            dtbuf, xc_f, xdbl, A_log + (size_t)i * D_INNER * N_STATE,
            Dvec + (size_t)i * D_INNER, ys);

        // gate: yz = ys * silu(z)
        mamba_gate<<<(MROWS * D_INNER + 255) / 256, 256, 0, stream>>>(ys, xz, yz_bf);

        // out_proj + residual: x = x + yz @ out_w^T (2048 x 768, K=1536) — NW=64
        cvt(out_proj_w + (size_t)i * D_MODEL * D_INNER, w_out,
            D_MODEL, D_INNER, D_MODEL, D_INNER, D_INNER);
        mamba_gemm_wmma<1, 64><<<dim3(D_MODEL / 64, M_CHUNKS), 256,
                                 64 * D_INNER * 2, stream>>>(
            yz_bf, w_out, x, MROWS, D_MODEL, D_INNER, x, M_CHUNK);
    }

    // final norm + lm_head: out = rms(x) @ lm_w^T (2048 x 50304, K=768) — NW=128
    mamba_rmsnorm<<<MROWS, 256, 0, stream>>>(x, xn, D_MODEL);
    cvt(lm_head_w, w_lm, VOCAB, D_MODEL, VOCAB, D_MODEL, D_MODEL);
    mamba_gemm_wmma<0, 128><<<dim3(VOCAB / 128, M_CHUNKS), 256,
                              128 * D_MODEL * 2, stream>>>(
        xn, w_lm, out, MROWS, VOCAB, D_MODEL, nullptr, M_CHUNK);
}
